// TransformerBlock_29308856828051
// MI455X (gfx1250) — matrix operand order repaired and checked by reading, not compiled, not run on hardware
//
#include <hip/hip_runtime.h>
#include <cmath>

// MI455X (gfx1250): full transformer block. All GEMMs via v_wmma_f32_16x16x32_f16
// (f16 operands, f32 accumulation), wave32, 16x64 output tile per wave.
// All B operands are in transposed (N x Kpad) layout -> contiguous 8B fragment
// loads. GEMM inner loop: single-exit counted loop over pairs of k-steps with
// the final pair peeled; loads issued one k-step ahead of their WMMAs so vmem
// latency overlaps the matrix pipe (partial s_wait_loadcnt, no acc phi copies).

#define TT   256          // T
#define EE   312          // E
#define DH   156          // DHEAD
#define BB   96
#define CC   5
#define KVK  1280         // C*V

typedef _Float16 v16h __attribute__((ext_vector_type(16)));
typedef _Float16 v8h  __attribute__((ext_vector_type(8)));
typedef _Float16 v4h  __attribute__((ext_vector_type(4)));
typedef float    v8f  __attribute__((ext_vector_type(8)));

__device__ __forceinline__ float gelu_f(float x) {
  return 0.5f * x * (1.0f + erff(x * 0.70710678118654752f));
}

__device__ __forceinline__ v16h ld16h(const _Float16* p) {
  v4h x0 = *(const v4h*)(p);     v4h x1 = *(const v4h*)(p + 4);
  v4h x2 = *(const v4h*)(p + 16); v4h x3 = *(const v4h*)(p + 20);
  v16h r;
  #pragma unroll
  for (int j = 0; j < 4; ++j) { r[j] = x0[j]; r[4+j] = x1[j]; r[8+j] = x2[j]; r[12+j] = x3[j]; }
  return r;
}
// A fragment: halves 0..7 at +0, halves 8..15 at +16 (16-bit A VGPR layout)
__device__ __forceinline__ v16h ld16h_a(const _Float16* p) {
  v4h x0 = *(const v4h*)(p);      v4h x1 = *(const v4h*)(p + 4);
  v4h x2 = *(const v4h*)(p + 16); v4h x3 = *(const v4h*)(p + 20);
  v16h r;
  #pragma unroll
  for (int j = 0; j < 4; ++j) { r[j] = x0[j]; r[4+j] = x1[j]; r[8+j] = x2[j]; r[12+j] = x3[j]; }
  return r;
}

// ------------------------------------------------ transpose-cast f32 -> f16
// dst[n][k] (row stride Kpad) = src[k][n]; zero-fills K/N padding so GEMM
// fragment loads over the padded region contribute exactly zero.
__global__ void k_cast_t(const float* __restrict__ src, _Float16* __restrict__ dst,
                         int K, int N, int Kpad, int Npad) {
  int i = blockIdx.x * 256 + threadIdx.x;
  if (i >= Kpad * Npad) return;
  int n = i / Kpad, k = i - n * Kpad;
  float v = (n < N && k < K) ? src[(size_t)k * N + n] : 0.f;
  dst[i] = (_Float16)v;
}

// ---------------------------------------------------------- sinogram gather
// X[b, t, c*V+v] = sino[b, c, r[t,v], c[t,v]] as f16. 4 consecutive k per
// thread: independent loads (clause-able, in flight together), 8B store.
__global__ void k_gather(const float* __restrict__ sino, const float* __restrict__ idx,
                         _Float16* __restrict__ X) {
  int i = blockIdx.x * 256 + threadIdx.x;
  const int total = BB * TT * KVK / 4;
  if (i >= total) return;
  int e0  = i * 4;
  int b   = e0 / (TT * KVK);
  int rem = e0 - b * (TT * KVK);
  int t   = rem / KVK;
  int k   = rem - t * KVK;
  v4h o;
  #pragma unroll
  for (int j = 0; j < 4; ++j) {
    int kk = k + j;
    int c = kk >> 8, v = kk & 255;
    int ib = (t * 256 + v) * 3;
    int ri = (int)idx[ib], ci = (int)idx[ib + 1];
    o[j] = (_Float16)sino[(((size_t)b * CC + c) << 16) + (ri << 8) + ci];
  }
  *(v4h*)(X + e0) = o;
}

// ------------------------------------------------------------ generic WMMA GEMM
// C[M=256 x N] = A(f16 row-major M x K, lda) @ Bt^T; Bt is N x K row-major.
// K must be a multiple of 64 (callers round up; weight Bts are zero-padded in K,
// so A's small over-read is multiplied by zero). TAILG adds one guarded A-tail
// step after the main loop (used by scores: K=156 -> KF=128 + 28 guarded).
// block=128 (4 waves): wave w -> M-tile blockIdx.y*4+w; blockIdx.x -> 64 cols.
// z decomposed (z/zdiv, z%zdiv) with independent strides (batch, head).
// EPI: 0 f16-store | 2 bias+gelu+f16 | 3 scale+f32 | 4 +res32->f32 | 5 bias+f32
//      | 6 f16 transposed store (out[col*ldo+row], vectorized)
//      | 7 bias + res32(row-indexed, e.g. pos) -> dual f32+f16 store
#define WMMA4(av, bv)                                                          \
  _Pragma("unroll")                                                            \
  for (int f = 0; f < 4; ++f)                                                  \
    acc[f] = __builtin_amdgcn_wmma_f32_16x16x32_f16(false, (av), false,        \
                                                    (bv)[f], (short)0, acc[f], \
                                                    false, false);

template <int EPI, bool TAILG>
__global__ __launch_bounds__(128) void k_gemm(
    const _Float16* __restrict__ A, int lda, long long aZ1, long long aZ2,
    const _Float16* __restrict__ Bt, int ldb, long long bZ1, long long bZ2,
    const float* __restrict__ bias,
    _Float16* __restrict__ out16, float* __restrict__ out32, int ldo,
    long long oZ1, long long oZ2,
    const float* __restrict__ res32, int ldr, long long rZ1, long long rZ2,
    float scale, int N, int K, int zdiv)
{
  const int z = blockIdx.z, z1 = z / zdiv, z2 = z % zdiv;
  A  += z1 * aZ1 + z2 * aZ2;
  Bt += z1 * bZ1 + z2 * bZ2;
  if (EPI == 0 || EPI == 2 || EPI == 6 || EPI == 7) out16 += z1 * oZ1 + z2 * oZ2;
  if (EPI == 3 || EPI == 4 || EPI == 5 || EPI == 7) out32 += z1 * oZ1 + z2 * oZ2;
  if (EPI == 4 || EPI == 7)                         res32 += z1 * rZ1 + z2 * rZ2;

  const int wave  = threadIdx.x >> 5, l = threadIdx.x & 31;
  const int mtile = blockIdx.y * 4 + wave;
  const int n0    = blockIdx.x * 4;
  const int arow  = mtile * 16 + (l & 15);
  const int akb   = (l < 16) ? 0 : 8;
  const int bkb   = (l < 16) ? 0 : 8;

  const int KF = TAILG ? (K & ~31) : K;       // KF is a multiple of 64

  const _Float16* ab = A + (size_t)arow * lda + akb;
  const _Float16* bb[4];
  #pragma unroll
  for (int f = 0; f < 4; ++f)
    bb[f] = Bt + (size_t)((n0 + f) * 16 + (l & 15)) * ldb + bkb;

  v8f acc[4] = {};

  // 2-stage software pipeline over pairs of 32-wide k-steps; last pair peeled
  // so the loop has a single exit (no accumulator phi copies).
  const int P = KF >> 6;                      // pairs, >= 2 for all stages
  v16h a0 = ld16h_a(ab);
  v16h b0[4], b1[4];
  #pragma unroll
  for (int f = 0; f < 4; ++f) b0[f] = ld16h(bb[f]);
  int k0 = 0;
  for (int it = 0; it < P - 1; ++it, k0 += 64) {
    v16h a1 = ld16h_a(ab + k0 + 32);
    #pragma unroll
    for (int f = 0; f < 4; ++f) b1[f] = ld16h(bb[f] + k0 + 32);
    __builtin_prefetch(ab + k0 + 128, 0, 3);  // speculative: drops on fault
    WMMA4(a0, b0)
    a0 = ld16h_a(ab + k0 + 64);
    #pragma unroll
    for (int f = 0; f < 4; ++f) b0[f] = ld16h(bb[f] + k0 + 64);
    WMMA4(a1, b1)
  }
  {                                           // final pair (k0 == KF - 64)
    v16h a1 = ld16h_a(ab + k0 + 32);
    v16h bl[4];
    #pragma unroll
    for (int f = 0; f < 4; ++f) bl[f] = ld16h(bb[f] + k0 + 32);
    WMMA4(a0, b0)
    WMMA4(a1, bl)
  }

  if (TAILG && (K & 31)) {                    // one guarded A step (A-side zeros)
    v16h a; v16h b[4];
    #pragma unroll
    for (int j = 0; j < 8; ++j) { int k = KF + akb + j;      a[j]   = (k < K) ? ab[KF + j]      : (_Float16)0.f; }
    #pragma unroll
    for (int j = 0; j < 8; ++j) { int k = KF + akb + 16 + j; a[8+j] = (k < K) ? ab[KF + 16 + j] : (_Float16)0.f; }
    #pragma unroll
    for (int f = 0; f < 4; ++f) b[f] = ld16h(bb[f] + KF);
    WMMA4(a, b)
  }

  const int r0 = mtile * 16 + ((l < 16) ? 0 : 8);
  #pragma unroll
  for (int f = 0; f < 4; ++f) {
    const int col = (n0 + f) * 16 + (l & 15);
    if (col >= N) continue;
    if (EPI == 6) {                            // transposed, vectorized f16 store
      v8h vs;
      #pragma unroll
      for (int j = 0; j < 8; ++j) vs[j] = (_Float16)acc[f][j];
      *(v8h*)(out16 + (size_t)col * ldo + r0) = vs;
      continue;
    }
    float bv = (EPI == 2 || EPI == 5 || EPI == 7) ? bias[col] : 0.f;
    #pragma unroll
    for (int j = 0; j < 8; ++j) {
      const int row = r0 + j;
      float v = acc[f][j];
      if (EPI == 2 || EPI == 5 || EPI == 7) v += bv;
      if (EPI == 2) v = gelu_f(v);
      if (EPI == 3) v *= scale;
      if (EPI == 4 || EPI == 7) v += res32[(size_t)row * ldr + col];
      if (EPI == 0 || EPI == 2) out16[(size_t)row * ldo + col] = (_Float16)v;
      else if (EPI == 7) { out32[(size_t)row * ldo + col] = v;
                           out16[(size_t)row * ldo + col] = (_Float16)v; }
      else              out32[(size_t)row * ldo + col] = v;
    }
  }
}

// --------------------------------------------------------------- row softmax
// 256 threads per score row; writes f16 att in-place into the first half of
// the row's own 1024B slot (row stride = 512 halfs).
__global__ __launch_bounds__(256) void k_softmax(float* __restrict__ scores) {
  const int row = blockIdx.x, tid = threadIdx.x;
  const int w = tid >> 5, l = tid & 31;
  __shared__ float sm[8], sv[8];
  float x = scores[(size_t)row * 256 + tid];

  float m = x;
  for (int o = 16; o > 0; o >>= 1) m = fmaxf(m, __shfl_xor(m, o, 32));
  if (l == 0) sm[w] = m;
  __syncthreads();
  float mm = sm[0];
  #pragma unroll
  for (int i = 1; i < 8; ++i) mm = fmaxf(mm, sm[i]);

  float e = __expf(x - mm);
  float s = e;
  for (int o = 16; o > 0; o >>= 1) s += __shfl_xor(s, o, 32);
  if (l == 0) sv[w] = s;
  __syncthreads();
  float ss = sv[0];
  #pragma unroll
  for (int i = 1; i < 8; ++i) ss += sv[i];

  _Float16* att = (_Float16*)scores;
  att[(size_t)row * 512 + tid] = (_Float16)(e / ss);   // barriers ordered all reads first
}

// ---------------------------------------------------------------- LayerNorm
__global__ __launch_bounds__(256) void k_layernorm(
    const float* __restrict__ in32, const _Float16* __restrict__ res16,
    const float* __restrict__ g, const float* __restrict__ beta,
    float* __restrict__ out32, _Float16* __restrict__ out16)
{
  const int row = blockIdx.x, tid = threadIdx.x;
  const size_t base = (size_t)row * EE;
  float xv[2]; int n = 0;
  float s = 0.f, sq = 0.f;
  for (int i = tid; i < EE; i += 256) {
    float x = in32[base + i];
    if (res16) x += (float)res16[base + i];
    xv[n++] = x; s += x; sq += x * x;
  }
  for (int o = 16; o > 0; o >>= 1) { s += __shfl_xor(s, o, 32); sq += __shfl_xor(sq, o, 32); }
  __shared__ float ss[8], ssq[8];
  const int w = tid >> 5, l = tid & 31;
  if (l == 0) { ss[w] = s; ssq[w] = sq; }
  __syncthreads();
  float ts = 0.f, tq = 0.f;
  #pragma unroll
  for (int i = 0; i < 8; ++i) { ts += ss[i]; tq += ssq[i]; }
  const float mean = ts / (float)EE;
  const float var  = tq / (float)EE - mean * mean;
  const float rstd = rsqrtf(var + 1e-5f);
  n = 0;
  for (int i = tid; i < EE; i += 256) {
    float y = (xv[n++] - mean) * rstd * g[i] + beta[i];
    if (out32) out32[base + i] = y;
    if (out16) out16[base + i] = (_Float16)y;
  }
}

// -------------------------------------------------------- scatter + residual
// Permutation covers every pixel exactly once:
// out[b, r, c] = sino[b, C-1, r, c] + O[b, t, v] * sc[t, v]
__global__ void k_scatter(const float* __restrict__ sino, const float* __restrict__ idx,
                          const float* __restrict__ O, float* __restrict__ out)
{
  int i = blockIdx.x * 256 + threadIdx.x;
  if (i >= BB * 256 * 256) return;
  int b = i >> 16, rem = i & 65535;
  int t = rem >> 8, v = rem & 255;
  int ib = (t * 256 + v) * 3;
  int ri = (int)idx[ib], ci = (int)idx[ib + 1];
  float sc = idx[ib + 2];
  int p = (ri << 8) + ci;
  out[((size_t)b << 16) + p] =
      sino[(((size_t)b * CC + (CC - 1)) << 16) + p] + O[(size_t)i] * sc;
}

// ============================================================== host launcher
extern "C" void kernel_launch(void* const* d_in, const int* in_sizes, int n_in,
                              void* d_out, int out_size, void* d_ws, size_t ws_size,
                              hipStream_t stream) {
  const float* sino    = (const float*)d_in[0];
  const float* indices = (const float*)d_in[1];
  const float* We      = (const float*)d_in[2];
  const float* be      = (const float*)d_in[3];
  const float* pos     = (const float*)d_in[4];
  const float* Wq      = (const float*)d_in[5];
  const float* Wk      = (const float*)d_in[6];
  const float* Wv      = (const float*)d_in[7];
  const float* W1      = (const float*)d_in[8];
  const float* b1      = (const float*)d_in[9];
  const float* W2      = (const float*)d_in[10];
  const float* b2      = (const float*)d_in[11];
  const float* W3      = (const float*)d_in[12];
  const float* b3      = (const float*)d_in[13];
  const float* g_att   = (const float*)d_in[14];
  const float* be_att  = (const float*)d_in[15];
  const float* g_mlp   = (const float*)d_in[16];
  const float* be_mlp  = (const float*)d_in[17];
  const float* Wu      = (const float*)d_in[18];
  const float* bu      = (const float*)d_in[19];
  float* out = (float*)d_out;

  const long long BTE = (long long)BB * TT * EE;   // 7,667,712
  const long long TE  = (long long)TT * EE;        // 79,872

  char* ws = (char*)d_ws;
  size_t off = 0;
  auto alloc = [&](size_t bytes) -> char* {
    char* p = ws + off;
    off = (off + bytes + 255) & ~(size_t)255;
    return p;
  };

  // transposed + padded f16 weights (N x Kpad row-major, zero padding)
  _Float16* W16  = (_Float16*)alloc(1105920 * sizeof(_Float16));
  _Float16* WeT  = W16;              // 320 x 1280
  _Float16* WqT  = W16 + 409600;     // 320 x 320
  _Float16* WkT  = W16 + 512000;
  _Float16* WvT  = W16 + 614400;
  _Float16* W1T  = W16 + 716800;
  _Float16* W2T  = W16 + 819200;
  _Float16* W3T  = W16 + 921600;
  _Float16* WuT  = W16 + 1024000;    // 256 x 320

  float*    buf0   = (float*)   alloc(BTE * 4);                      // Z f32 -> Z+a -> x1
  _Float16* buf1   = (_Float16*)alloc(BTE * 2);                      // Z f16 -> x1 f16
  _Float16* Q16    = (_Float16*)alloc(BTE * 2);
  _Float16* K16    = (_Float16*)alloc(BTE * 2);
  _Float16* VT16   = (_Float16*)alloc(BTE * 2);                      // V transposed (B, E, T)
  float*    scores = (float*)   alloc((size_t)BB * 2 * TT * TT * 4); // + att f16 in-place + final O
  _Float16* buf3   = (_Float16*)alloc(BTE * 2);                      // MLP ping
  _Float16* buf4   = (_Float16*)alloc(BTE * 2);                      // MLP pong / x2 f16
  (void)alloc(256);                                                  // tail pad (A over-read, K=320)
  _Float16* att16  = (_Float16*)scores;                              // row stride 512 halfs
  float*    Ofs    = scores;                                         // final projection output
  // Gathered X (B,T,1280) f16 = 62.9MB: aliases Q16..scores, which are only
  // written after the embed GEMM has consumed X (stream-ordered).
  _Float16* X16    = Q16;
  (void)ws_size; (void)in_sizes; (void)n_in; (void)out_size;

  // 1) transpose-cast weights to f16 (zero-padded)
  struct { const float* s; _Float16* d; int K, N, Kp, Np; } tc[8] = {
    {We, WeT, 1280, 312, 1280, 320}, {Wq, WqT, 312, 312, 320, 320},
    {Wk, WkT, 312, 312, 320, 320},   {Wv, WvT, 312, 312, 320, 320},
    {W1, W1T, 312, 312, 320, 320},   {W2, W2T, 312, 312, 320, 320},
    {W3, W3T, 312, 312, 320, 320},   {Wu, WuT, 312, 256, 320, 256}};
  for (int i = 0; i < 8; ++i) {
    int n = tc[i].Kp * tc[i].Np;
    k_cast_t<<<(n + 255) / 256, 256, 0, stream>>>(tc[i].s, tc[i].d, tc[i].K, tc[i].N, tc[i].Kp, tc[i].Np);
  }

  // 2) gather X = sino[:, :, r, c] (f16, streaming)
  k_gather<<<(BB * TT * KVK / 4 + 255) / 256, 256, 0, stream>>>(sino, indices, X16);

  // 3) embed GEMM: Z = X@We + be + pos -> f32 (buf0) + f16 (buf1)
  k_gemm<7, false><<<dim3(5, 4, BB), 128, 0, stream>>>(
      X16, KVK, (long long)TT * KVK, 0, WeT, KVK, 0, 0, be,
      buf1, buf0, EE, TE, 0,
      pos, EE, 0, 0, 1.f, EE, KVK, 1);

  // 4) Q/K = Z @ Wq/Wk ; V stored transposed (K run to 320: B zero-padded)
  k_gemm<0, false><<<dim3(5, 4, BB), 128, 0, stream>>>(
      buf1, EE, TE, 0, WqT, 320, 0, 0, nullptr, Q16, nullptr, EE, TE, 0,
      nullptr, 0, 0, 0, 1.f, EE, 320, 1);
  k_gemm<0, false><<<dim3(5, 4, BB), 128, 0, stream>>>(
      buf1, EE, TE, 0, WkT, 320, 0, 0, nullptr, K16, nullptr, EE, TE, 0,
      nullptr, 0, 0, 0, 1.f, EE, 320, 1);
  k_gemm<6, false><<<dim3(5, 4, BB), 128, 0, stream>>>(
      buf1, EE, TE, 0, WvT, 320, 0, 0, nullptr, VT16, nullptr, TT, TE, 0,
      nullptr, 0, 0, 0, 1.f, EE, 320, 1);

  // 5) scores = Q @ K^T * DHEAD^-0.5   (z = b*2 + h; K16 rows serve as Bt)
  const float sscale = 1.0f / sqrtf((float)DH);
  k_gemm<3, true><<<dim3(4, 4, BB * 2), 128, 0, stream>>>(
      Q16, EE, TE, DH, K16, EE, TE, DH, nullptr,
      nullptr, scores, TT, 2LL * TT * TT, (long long)TT * TT,
      nullptr, 0, 0, 0, sscale, TT, DH, 2);

  // 6) softmax rows -> att f16 (in-place row slots)
  k_softmax<<<BB * 2 * TT, 256, 0, stream>>>(scores);

  // 7) x1pre = Z + att @ V  (f32, in-place residual into buf0); B = VT
  k_gemm<4, false><<<dim3(3, 4, BB * 2), 128, 0, stream>>>(
      att16, 512, 2LL * TT * 512, (long long)TT * 512,
      VT16, TT, TE, (long long)DH * TT, nullptr,
      nullptr, buf0, EE, TE, DH,
      buf0, EE, TE, DH, 1.f, DH, TT, 2);

  // 8) LN1: x1 (f32 in buf0, f16 in buf1)
  k_layernorm<<<BB * TT, 256, 0, stream>>>(buf0, nullptr, g_att, be_att, buf0, buf1);

  // 9) MLP: three bias+GELU GEMMs (K run to 320: B zero-padded)
  k_gemm<2, false><<<dim3(5, 4, BB), 128, 0, stream>>>(
      buf1, EE, TE, 0, W1T, 320, 0, 0, b1, buf3, nullptr, EE, TE, 0,
      nullptr, 0, 0, 0, 1.f, EE, 320, 1);
  k_gemm<2, false><<<dim3(5, 4, BB), 128, 0, stream>>>(
      buf3, EE, TE, 0, W2T, 320, 0, 0, b2, buf4, nullptr, EE, TE, 0,
      nullptr, 0, 0, 0, 1.f, EE, 320, 1);
  k_gemm<2, false><<<dim3(5, 4, BB), 128, 0, stream>>>(
      buf4, EE, TE, 0, W3T, 320, 0, 0, b3, buf3, nullptr, EE, TE, 0,
      nullptr, 0, 0, 0, 1.f, EE, 320, 1);

  // 10) LN2: x2 = LN(x1 + h) -> f16 buf4
  k_layernorm<<<BB * TT, 256, 0, stream>>>(buf0, buf3, g_mlp, be_mlp, nullptr, buf4);

  // 11) out = x2 @ Wu + bu -> f32 (reuses scores region)
  k_gemm<5, false><<<dim3(4, 4, BB), 128, 0, stream>>>(
      buf4, EE, TE, 0, WuT, 320, 0, 0, bu,
      nullptr, Ofs, TT, (long long)TT * TT, 0,
      nullptr, 0, 0, 0, 1.f, TT, 320, 1);

  // 12) scatter through permutation onto last sino channel
  k_scatter<<<(BB * 256 * 256 + 255) / 256, 256, 0, stream>>>(sino, indices, Ofs, out);
}
